// GAT_91139206021463
// MI455X (gfx1250) — matrix operand order repaired and checked by reading, not compiled, not run on hardware
//
#include <hip/hip_runtime.h>
#include <hip/hip_bf16.h>
#include <cstdint>

// ---------------- types ----------------
typedef __attribute__((ext_vector_type(16))) __bf16 v16bf;
typedef __attribute__((ext_vector_type(8)))  __bf16 v8bf;
typedef __attribute__((ext_vector_type(8)))  float  v8f;

#define Bsz 32
#define Ssz 1024
#define Dsz 768
#define Hsz 12
#define DHsz 64
#define ROWS (Bsz * Ssz)          // 32768
#define NEGV (-1000000000.0f)

__device__ __forceinline__ unsigned short f2bf(float f) {
    union { float f; unsigned u; } v; v.f = f;
    unsigned r = v.u + 0x7fffu + ((v.u >> 16) & 1u);
    return (unsigned short)(r >> 16);
}
__device__ __forceinline__ float bf2f(unsigned short h) {
    union { unsigned u; float f; } v; v.u = ((unsigned)h) << 16; return v.f;
}

// async 16B global -> LDS copy (CDNA5, ASYNCcnt-tracked), per-lane addresses
__device__ __forceinline__ void async_cp16(unsigned lds_off, const void* gptr) {
    asm volatile("global_load_async_to_lds_b128 %0, %1, off"
                 :: "v"(lds_off), "v"((unsigned long long)(uintptr_t)gptr)
                 : "memory");
}
__device__ __forceinline__ void wait_async0() {
    asm volatile("s_wait_asynccnt 0x0" ::: "memory");
}

// ---------------- WMMA GEMM: C(MxN) = A(MxK,bf16) * Bt(NxK,bf16)^T ----------------
// A row-major (ldA), Bt N-major with contiguous K (ldB). Block tile 128x64, BK=32,
// 256 threads (8 waves, 4x2), each wave a 32x32 strip (4 accumulators).
// Double-buffered LDS fed by global_load_async_to_lds_b128.
// mode 0: store bf16   mode 1: bias+relu -> bf16   mode 2: Cf += acc (fp32)
#define LDSS 40
__global__ __launch_bounds__(256)
void gemm_bf16_wmma(const unsigned short* __restrict__ A,
                    const unsigned short* __restrict__ Bt,
                    unsigned short* __restrict__ Cbf,
                    float* __restrict__ Cf,
                    const float* __restrict__ bias,
                    int K, int ldA, int ldB, int ldC, int mode)
{
    __shared__ __align__(16) unsigned short As[2][128 * LDSS];
    __shared__ __align__(16) unsigned short Bs[2][64 * LDSS];
    const int tid  = threadIdx.x;
    const int m0   = blockIdx.y * 128;
    const int n0   = blockIdx.x * 64;
    const int wave = tid >> 5, lane = tid & 31;
    const int wm   = (wave >> 1) * 32;   // wave M base within block (0/32/64/96)
    const int wn   = (wave & 1) * 32;    // wave N base within block (0/32)

    v8f acc[2][2] = {};

    // tile copy mapping: A = 512 16B chunks (2/thread), B = 256 chunks (1/thread)
    const int ar0 = (tid * 2) >> 2, as0 = (tid * 2) & 3;
    const int ar1 = (tid * 2 + 1) >> 2, as1 = (tid * 2 + 1) & 3;
    const int br = tid >> 2, bs = tid & 3;

    auto load_tile = [&](int k0, int b) {
        async_cp16((unsigned)(uintptr_t)&As[b][ar0 * LDSS + as0 * 8],
                   &A[(size_t)(m0 + ar0) * ldA + k0 + as0 * 8]);
        async_cp16((unsigned)(uintptr_t)&As[b][ar1 * LDSS + as1 * 8],
                   &A[(size_t)(m0 + ar1) * ldA + k0 + as1 * 8]);
        async_cp16((unsigned)(uintptr_t)&Bs[b][br * LDSS + bs * 8],
                   &Bt[(size_t)(n0 + br) * ldB + k0 + bs * 8]);
    };

    int buf = 0;
    load_tile(0, 0);
    wait_async0();
    __syncthreads();

    for (int k0 = 0; k0 < K; k0 += 32) {
        if (k0 + 32 < K) load_tile(k0 + 32, buf ^ 1);   // DMA next tile (overlaps WMMA)

        const int kb   = (lane >> 4) * 8;    // A frag layout: K {kb..kb+7, 16+kb..}
        const int koff = (lane >> 4) * 16;   // B frag layout: same as A (K {kb..kb+7, 16+kb..})
        const int l15  = lane & 15;

        v16bf afrag[2], bfrag[2];
        #pragma unroll
        for (int mi = 0; mi < 2; ++mi) {
            const unsigned short* p = &As[buf][(wm + mi * 16 + l15) * LDSS];
            v8bf lo = *reinterpret_cast<const v8bf*>(p + kb);
            v8bf hi = *reinterpret_cast<const v8bf*>(p + 16 + kb);
            #pragma unroll
            for (int i = 0; i < 8; ++i) { afrag[mi][i] = lo[i]; afrag[mi][i + 8] = hi[i]; }
        }
        #pragma unroll
        for (int ni = 0; ni < 2; ++ni) {
            const unsigned short* p = &Bs[buf][(wn + ni * 16 + l15) * LDSS];
            v8bf lo = *reinterpret_cast<const v8bf*>(p + kb);
            v8bf hi = *reinterpret_cast<const v8bf*>(p + 16 + kb);
            #pragma unroll
            for (int i = 0; i < 8; ++i) { bfrag[ni][i] = lo[i]; bfrag[ni][i + 8] = hi[i]; }
        }
        #pragma unroll
        for (int mi = 0; mi < 2; ++mi)
            #pragma unroll
            for (int ni = 0; ni < 2; ++ni)
                acc[mi][ni] = __builtin_amdgcn_wmma_f32_16x16x32_bf16(
                    false, afrag[mi], false, bfrag[ni], (short)0, acc[mi][ni], false, false);

        wait_async0();
        __syncthreads();
        buf ^= 1;
    }

    // epilogue: C layout lanes 0-15 -> M=r, lanes 16-31 -> M=r+8 ; N = lane&15
    const int colBase = n0 + wn + (lane & 15);
    const int rowBase = m0 + wm + ((lane >> 4) << 3);
    #pragma unroll
    for (int mi = 0; mi < 2; ++mi) {
        #pragma unroll
        for (int ni = 0; ni < 2; ++ni) {
            const int col = colBase + ni * 16;
            #pragma unroll
            for (int r = 0; r < 8; ++r) {
                const size_t off = (size_t)(rowBase + mi * 16 + r) * ldC + col;
                float v = acc[mi][ni][r];
                if (mode == 0) {
                    Cbf[off] = f2bf(v);
                } else if (mode == 1) {
                    v += bias[col];
                    Cbf[off] = f2bf(v > 0.f ? v : 0.f);
                } else {
                    Cf[off] += v;
                }
            }
        }
    }
}

// ---------------- helpers / small kernels ----------------
__global__ void cast_bf16_kernel(const float* __restrict__ in,
                                 unsigned short* __restrict__ out, int n) {
    int i = blockIdx.x * 256 + threadIdx.x;
    if (i < n) out[i] = f2bf(in[i]);
}

// Wc (H,D,DH) -> N-major (H*DH rows, D cols) bf16: out[n*D + i] = Wc[h,i,d], n=h*64+d
__global__ void wc_cat_t_kernel(const float* __restrict__ Wc,
                                unsigned short* __restrict__ out) {
    int idx = blockIdx.x * 256 + threadIdx.x;
    if (idx >= Dsz * Dsz) return;
    int n = idx / Dsz, i = idx % Dsz, h = n >> 6, d = n & 63;
    out[idx] = f2bf(Wc[(size_t)h * Dsz * DHsz + (size_t)i * DHsz + d]);
}

// generic transpose-cast: in (R x C) fp32 -> out (C x R) bf16
__global__ void transpose_cast_kernel(const float* __restrict__ in,
                                      unsigned short* __restrict__ out, int R, int C) {
    int idx = blockIdx.x * 256 + threadIdx.x;
    if (idx >= R * C) return;
    int r = idx / C, c = idx % C;
    out[(size_t)c * R + r] = f2bf(in[idx]);
}

// attn vecs: v0=Wt@Wa[:,:64] v1=Wt@Wa1[:,:64] v2=Wg@Wa[:,64:] v3=Wg@Wa1[:,64:]
__global__ void attn_vec_kernel(const float* __restrict__ Wt, const float* __restrict__ Wg,
                                const float* __restrict__ Wa, const float* __restrict__ Wa1,
                                float* __restrict__ out) {
    int idx = blockIdx.x * 256 + threadIdx.x;
    if (idx >= 4 * Hsz * Dsz) return;
    int v = idx / (Hsz * Dsz);
    int h = (idx / Dsz) % Hsz;
    int i = idx % Dsz;
    const float* W  = (v < 2) ? Wt : Wg;
    const float* av = (v & 1) ? Wa1 : Wa;
    int off = (v < 2) ? 0 : DHsz;
    float acc = 0.f;
    #pragma unroll 4
    for (int d = 0; d < DHsz; ++d)
        acc += W[(size_t)h * Dsz * DHsz + (size_t)i * DHsz + d] * av[h * 2 * DHsz + off + d];
    out[idx] = acc;
}

// per (b,h): tt = topic_row . vec  (both Wa and Wa1 variants)
__global__ __launch_bounds__(32)
void topic_term_kernel(const float* __restrict__ src,
                       const float* __restrict__ wt_a, const float* __restrict__ wt_a1,
                       float* __restrict__ tt0, float* __restrict__ tt1) {
    int b = blockIdx.x, h = blockIdx.y, lane = threadIdx.x;
    const float* row = src + (size_t)b * Dsz;
    float a0 = 0.f, a1 = 0.f;
    for (int i = lane; i < Dsz; i += 32) {
        float x = row[i];
        a0 += x * wt_a[h * Dsz + i];
        a1 += x * wt_a1[h * Dsz + i];
    }
    for (int o = 16; o > 0; o >>= 1) {
        a0 += __shfl_down(a0, o, 32);
        a1 += __shfl_down(a1, o, 32);
    }
    if (lane == 0) { tt0[b * Hsz + h] = a0; tt1[b * Hsz + h] = a1; }
}

// per (b,s): z[b,h,s] for both variants; leaky_relu + mask
__global__ __launch_bounds__(128)
void compute_z_kernel(const unsigned short* __restrict__ Abf,
                      const float* __restrict__ wg_a, const float* __restrict__ wg_a1,
                      const float* __restrict__ tt0, const float* __restrict__ tt1,
                      const int* __restrict__ mask,
                      float* __restrict__ z0, float* __restrict__ z1) {
    __shared__ float row[Dsz];
    const int bs = blockIdx.x;
    const int b = bs >> 10, s = bs & 1023;
    const int tid = threadIdx.x;
    for (int i = tid; i < Dsz; i += 128) row[i] = bf2f(Abf[(size_t)bs * Dsz + i]);
    __syncthreads();
    const int wave = tid >> 5, lane = tid & 31;
    const bool masked = mask[bs] != 0;
    for (int r = 0; r < 6; ++r) {
        const int j = wave * 6 + r;
        const int h = j >> 1, v = j & 1;
        const float* vec = (v ? wg_a1 : wg_a) + h * Dsz;
        float acc = 0.f;
        for (int i = lane; i < Dsz; i += 32) acc += row[i] * vec[i];
        for (int o = 16; o > 0; o >>= 1) acc += __shfl_down(acc, o, 32);
        if (lane == 0) {
            float zz = (v ? tt1 : tt0)[b * Hsz + h] + acc;
            zz = zz > 0.f ? zz : 0.01f * zz;       // leaky_relu
            if (masked) zz = NEGV;
            (v ? z1 : z0)[((size_t)(b * Hsz + h) << 10) + s] = zz;
        }
    }
}

// softmax over S per (b,h) row, in place; blockIdx.y = variant
__global__ __launch_bounds__(256)
void softmax_kernel(float* __restrict__ z0, float* __restrict__ z1) {
    float* p = (blockIdx.y ? z1 : z0) + ((size_t)blockIdx.x << 10);
    __shared__ float red[256];
    const int tid = threadIdx.x;
    float v[4], m = -3.4e38f;
    #pragma unroll
    for (int i = 0; i < 4; ++i) { v[i] = p[tid + 256 * i]; m = fmaxf(m, v[i]); }
    red[tid] = m; __syncthreads();
    for (int o = 128; o > 0; o >>= 1) { if (tid < o) red[tid] = fmaxf(red[tid], red[tid + o]); __syncthreads(); }
    m = red[0]; __syncthreads();
    float s = 0.f;
    #pragma unroll
    for (int i = 0; i < 4; ++i) { v[i] = __expf(v[i] - m); s += v[i]; }
    red[tid] = s; __syncthreads();
    for (int o = 128; o > 0; o >>= 1) { if (tid < o) red[tid] += red[tid + o]; __syncthreads(); }
    float inv = 1.0f / red[0];
    #pragma unroll
    for (int i = 0; i < 4; ++i) p[tid + 256 * i] = v[i] * inv;
}

// out[b,s,n] = tanh(a[b, n>>6, s] * fcg[b,s,n])
__global__ void attn_tanh_kernel(const float* __restrict__ a,
                                 const unsigned short* __restrict__ fcg,
                                 float* __restrict__ out, int total) {
    int idx = blockIdx.x * 256 + threadIdx.x;
    if (idx >= total) return;
    int n = idx % Dsz;
    int row = idx / Dsz;
    int b = row >> 10, s = row & 1023, h = n >> 6;
    float av = a[((size_t)(b * Hsz + h) << 10) + s];
    out[idx] = tanhf(av * bf2f(fcg[idx]));
}

// t[b, h*64+d] = tanh( sum_s a1[b,h,s] * fcg[b,s,h*64+d] )
__global__ __launch_bounds__(256)
void topic_reduce_kernel(const float* __restrict__ a1,
                         const unsigned short* __restrict__ fcg,
                         float* __restrict__ tout) {
    const int b = blockIdx.x / Hsz, h = blockIdx.x % Hsz;
    const int tid = threadIdx.x;
    const int d = tid & 63, g = tid >> 6;
    __shared__ float red[256];
    const float* ap = a1 + ((size_t)(b * Hsz + h) << 10);
    float acc = 0.f;
    for (int s = g; s < Ssz; s += 4)
        acc += ap[s] * bf2f(fcg[((size_t)(b * Ssz + s)) * Dsz + h * DHsz + d]);
    red[tid] = acc; __syncthreads();
    if (tid < 64) {
        float v = red[d] + red[d + 64] + red[d + 128] + red[d + 192];
        tout[(size_t)b * Dsz + h * DHsz + d] = tanhf(v);
    }
}

// layernorm over D=768, out bf16
__global__ __launch_bounds__(256)
void ln_bf16_kernel(const float* __restrict__ x, unsigned short* __restrict__ out,
                    const float* __restrict__ g, const float* __restrict__ bta, float eps) {
    const size_t row = blockIdx.x;
    const float* p = x + row * Dsz;
    const int tid = threadIdx.x;
    float v0 = p[tid], v1 = p[tid + 256], v2 = p[tid + 512];
    __shared__ float red[256];
    red[tid] = v0 + v1 + v2; __syncthreads();
    for (int o = 128; o > 0; o >>= 1) { if (tid < o) red[tid] += red[tid + o]; __syncthreads(); }
    float mean = red[0] * (1.f / 768.f); __syncthreads();
    float d0 = v0 - mean, d1 = v1 - mean, d2 = v2 - mean;
    red[tid] = d0 * d0 + d1 * d1 + d2 * d2; __syncthreads();
    for (int o = 128; o > 0; o >>= 1) { if (tid < o) red[tid] += red[tid + o]; __syncthreads(); }
    float rstd = rsqrtf(red[0] * (1.f / 768.f) + eps);
    unsigned short* op = out + row * Dsz;
    op[tid]       = f2bf(d0 * rstd * g[tid]       + bta[tid]);
    op[tid + 256] = f2bf(d1 * rstd * g[tid + 256] + bta[tid + 256]);
    op[tid + 512] = f2bf(d2 * rstd * g[tid + 512] + bta[tid + 512]);
}

// layernorm, fp32 out (small row counts)
__global__ __launch_bounds__(256)
void ln_f32_kernel(const float* __restrict__ x, float* __restrict__ out,
                   const float* __restrict__ g, const float* __restrict__ bta, float eps) {
    const size_t row = blockIdx.x;
    const float* p = x + row * Dsz;
    const int tid = threadIdx.x;
    float v0 = p[tid], v1 = p[tid + 256], v2 = p[tid + 512];
    __shared__ float red[256];
    red[tid] = v0 + v1 + v2; __syncthreads();
    for (int o = 128; o > 0; o >>= 1) { if (tid < o) red[tid] += red[tid + o]; __syncthreads(); }
    float mean = red[0] * (1.f / 768.f); __syncthreads();
    float d0 = v0 - mean, d1 = v1 - mean, d2 = v2 - mean;
    red[tid] = d0 * d0 + d1 * d1 + d2 * d2; __syncthreads();
    for (int o = 128; o > 0; o >>= 1) { if (tid < o) red[tid] += red[tid + o]; __syncthreads(); }
    float rstd = rsqrtf(red[0] * (1.f / 768.f) + eps);
    float* op = out + row * Dsz;
    op[tid]       = d0 * rstd * g[tid]       + bta[tid];
    op[tid + 256] = d1 * rstd * g[tid + 256] + bta[tid + 256];
    op[tid + 512] = d2 * rstd * g[tid + 512] + bta[tid + 512];
}

__global__ void residual_kernel(const float* __restrict__ x, const float* __restrict__ b2,
                                float* __restrict__ out, int total) {
    int idx = blockIdx.x * 256 + threadIdx.x;
    if (idx < total) out[idx] = x[idx] + b2[idx % Dsz];
}

__global__ void broadcast_kernel(const float* __restrict__ rows, float* __restrict__ out, int total) {
    int idx = blockIdx.x * 256 + threadIdx.x;
    if (idx >= total) return;
    int b = idx / (Ssz * Dsz), n = idx % Dsz;
    out[idx] = rows[b * Dsz + n];
}

// small FFN on topic rows (32 rows), fp32
__global__ void small_gemm_relu_kernel(const float* __restrict__ h, const float* __restrict__ w1,
                                       const float* __restrict__ b1, float* __restrict__ inter) {
    int b = blockIdx.x, f = blockIdx.y * 256 + threadIdx.x;
    float acc = 0.f;
    for (int i = 0; i < Dsz; ++i) acc += h[b * Dsz + i] * w1[(size_t)i * (3 * Dsz) + f];
    acc += b1[f];
    inter[b * (3 * Dsz) + f] = acc > 0.f ? acc : 0.f;
}
__global__ void small_gemm_out_kernel(const float* __restrict__ inter, const float* __restrict__ w2,
                                      const float* __restrict__ b2, const float* __restrict__ t2,
                                      float* __restrict__ outt) {
    int b = blockIdx.x, n = blockIdx.y * 256 + threadIdx.x;
    float acc = 0.f;
    for (int f = 0; f < 3 * Dsz; ++f) acc += inter[b * (3 * Dsz) + f] * w2[(size_t)f * Dsz + n];
    outt[b * Dsz + n] = acc + b2[n] + t2[b * Dsz + n];
}

// ---------------- launch ----------------
extern "C" void kernel_launch(void* const* d_in, const int* in_sizes, int n_in,
                              void* d_out, int out_size, void* d_ws, size_t ws_size,
                              hipStream_t stream) {
    const float* topic = (const float*)d_in[0];
    const float* gce   = (const float*)d_in[1];
    const int*   mask  = (const int*)d_in[2];
    const float* Wt    = (const float*)d_in[3];
    const float* Wg    = (const float*)d_in[4];
    const float* Wc    = (const float*)d_in[5];
    const float* Wa    = (const float*)d_in[6];
    const float* Wa1   = (const float*)d_in[7];
    const float* ng    = (const float*)d_in[8];
    const float* nb    = (const float*)d_in[9];
    const float* pwg   = (const float*)d_in[10];
    const float* pwb   = (const float*)d_in[11];
    const float* w1    = (const float*)d_in[12];
    const float* b1    = (const float*)d_in[13];
    const float* w2    = (const float*)d_in[14];
    const float* b2    = (const float*)d_in[15];

    float* O0 = (float*)d_out;                    // out_gl (also scratch: gce1)
    float* O1 = O0 + (size_t)ROWS * Dsz;          // out_tp (also scratch: gce2)

    // workspace carve-out
    char* w = (char*)d_ws;
    auto alloc = [&](size_t bytes) -> char* {
        char* p = w; w += (bytes + 255) & ~(size_t)255; return p;
    };
    unsigned short* A_bf = (unsigned short*)alloc((size_t)ROWS * Dsz * 2);
    unsigned short* F_bf = (unsigned short*)alloc((size_t)ROWS * Dsz * 2);
    unsigned short* wc_t = (unsigned short*)alloc((size_t)Dsz * Dsz * 2);       // (768 x 768) N-major
    unsigned short* w1t  = (unsigned short*)alloc((size_t)3 * Dsz * Dsz * 2);   // (2304 x 768)
    unsigned short* w2t  = (unsigned short*)alloc((size_t)Dsz * 3 * Dsz * 2);   // (768 x 2304)
    float* avecs = (float*)alloc((size_t)4 * Hsz * Dsz * 4);
    float* tt0   = (float*)alloc((size_t)Bsz * Hsz * 4);
    float* tt1   = (float*)alloc((size_t)Bsz * Hsz * 4);
    float* z0    = (float*)alloc((size_t)Bsz * Hsz * Ssz * 4);
    float* z1    = (float*)alloc((size_t)Bsz * Hsz * Ssz * 4);
    float* t1    = (float*)alloc((size_t)Bsz * Dsz * 4);
    float* lt    = (float*)alloc((size_t)Bsz * Dsz * 4);
    float* t2    = (float*)alloc((size_t)Bsz * Dsz * 4);
    float* ht    = (float*)alloc((size_t)Bsz * Dsz * 4);
    float* intert = (float*)alloc((size_t)Bsz * 3 * Dsz * 4);
    float* outt  = (float*)alloc((size_t)Bsz * Dsz * 4);

    const float* wt_a  = avecs;
    const float* wt_a1 = avecs + 1 * Hsz * Dsz;
    const float* wg_a  = avecs + 2 * Hsz * Dsz;
    const float* wg_a1 = avecs + 3 * Hsz * Dsz;

    const int total = ROWS * Dsz;                 // 25,165,824
    const dim3 gemm_grid(Dsz / 64, ROWS / 128);   // (12, 256)

    // --- prep ---
    cast_bf16_kernel<<<(total + 255) / 256, 256, 0, stream>>>(gce, A_bf, total);
    wc_cat_t_kernel<<<(Dsz * Dsz + 255) / 256, 256, 0, stream>>>(Wc, wc_t);
    transpose_cast_kernel<<<(Dsz * 3 * Dsz + 255) / 256, 256, 0, stream>>>(w1, w1t, Dsz, 3 * Dsz);
    transpose_cast_kernel<<<(3 * Dsz * Dsz + 255) / 256, 256, 0, stream>>>(w2, w2t, 3 * Dsz, Dsz);
    attn_vec_kernel<<<(4 * Hsz * Dsz + 255) / 256, 256, 0, stream>>>(Wt, Wg, Wa, Wa1, avecs);

    // --- multihead call 1 ---
    topic_term_kernel<<<dim3(Bsz, Hsz), 32, 0, stream>>>(topic, wt_a, wt_a1, tt0, tt1);
    gemm_bf16_wmma<<<gemm_grid, 256, 0, stream>>>(A_bf, wc_t, F_bf, nullptr, nullptr,
                                                  Dsz, Dsz, Dsz, Dsz, 0);          // fcg1
    compute_z_kernel<<<ROWS, 128, 0, stream>>>(A_bf, wg_a, wg_a1, tt0, tt1, mask, z0, z1);
    softmax_kernel<<<dim3(Bsz * Hsz, 2), 256, 0, stream>>>(z0, z1);
    attn_tanh_kernel<<<(total + 255) / 256, 256, 0, stream>>>(z0, F_bf, O0, total); // gce1
    topic_reduce_kernel<<<Bsz * Hsz, 256, 0, stream>>>(z1, F_bf, t1);

    // --- multihead call 2 ---
    ln_bf16_kernel<<<ROWS, 256, 0, stream>>>(O0, A_bf, ng, nb, 1e-5f);   // LN(gce1) bf16
    ln_f32_kernel<<<Bsz, 256, 0, stream>>>(t1, lt, ng, nb, 1e-5f);       // LN(t1)
    topic_term_kernel<<<dim3(Bsz, Hsz), 32, 0, stream>>>(lt, wt_a, wt_a1, tt0, tt1);
    gemm_bf16_wmma<<<gemm_grid, 256, 0, stream>>>(A_bf, wc_t, F_bf, nullptr, nullptr,
                                                  Dsz, Dsz, Dsz, Dsz, 0);          // fcg2
    compute_z_kernel<<<ROWS, 128, 0, stream>>>(A_bf, wg_a, wg_a1, tt0, tt1, mask, z0, z1);
    softmax_kernel<<<dim3(Bsz * Hsz, 2), 256, 0, stream>>>(z0, z1);
    attn_tanh_kernel<<<(total + 255) / 256, 256, 0, stream>>>(z0, F_bf, O1, total); // gce2
    topic_reduce_kernel<<<Bsz * Hsz, 256, 0, stream>>>(z1, F_bf, t2);

    // --- FFN (out_gl) ---
    ln_bf16_kernel<<<ROWS, 256, 0, stream>>>(O1, A_bf, pwg, pwb, 1e-6f); // h2 bf16
    residual_kernel<<<(total + 255) / 256, 256, 0, stream>>>(O1, b2, O0, total);

    // --- FFN (out_tp) on 32 unique rows, then broadcast into O1 ---
    ln_f32_kernel<<<Bsz, 256, 0, stream>>>(t2, ht, pwg, pwb, 1e-6f);
    small_gemm_relu_kernel<<<dim3(Bsz, (3 * Dsz) / 256), 256, 0, stream>>>(ht, w1, b1, intert);
    small_gemm_out_kernel<<<dim3(Bsz, Dsz / 256), 256, 0, stream>>>(intert, w2, b2, t2, outt);
    broadcast_kernel<<<(total + 255) / 256, 256, 0, stream>>>(outt, O1, total);

    // --- big FFN GEMMs, 3 K-chunks of the 2304 inter dim ---
    for (int c = 0; c < 3; ++c) {
        gemm_bf16_wmma<<<gemm_grid, 256, 0, stream>>>(
            A_bf, w1t + (size_t)c * Dsz * Dsz, F_bf, nullptr, b1 + c * Dsz,
            Dsz, Dsz, Dsz, Dsz, 1);                       // inter chunk, relu+bias -> bf16
        gemm_bf16_wmma<<<gemm_grid, 256, 0, stream>>>(
            F_bf, w2t + c * Dsz, nullptr, O0, nullptr,
            Dsz, Dsz, 3 * Dsz, Dsz, 2);                   // O0 += chunk @ w2_c
    }
}
